// SpineSegmentationNet_42253888258734
// MI455X (gfx1250) — matrix operand order repaired and checked by reading, not compiled, not run on hardware
//
#include <hip/hip_runtime.h>
#include <hip/hip_bf16.h>
#include <math.h>

// ---------------------------------------------------------------------------
// CDNA5 (gfx1250) implementation of the PointNet++-MSG segmentation network.
// Matrix work runs on v_wmma_f32_16x16x32_f16 (f16 operands, f32 accum) —
// per-point MLP GEMMs feed BatchNorm, so f16 operand precision is ample and
// f32 accumulation preserves channel statistics. LDS tiles are stored in the
// WMMA fragment layout so each lane fetches its v16h operand with two
// ds_load_b128 ops; tile loads are software-pipelined (regs hold tile k+1
// while tile k is consumed from LDS).
// ---------------------------------------------------------------------------

typedef __attribute__((ext_vector_type(16))) _Float16 v16h;
typedef __attribute__((ext_vector_type(8)))  float    v8f;

static __device__ __forceinline__ unsigned pack2h(float a, float b)
{
    union { _Float16 h[2]; unsigned u; } x;
    x.h[0] = (_Float16)a;
    x.h[1] = (_Float16)b;
    return x.u;
}

// Per-thread registers for one 64x32 A tile + 32x64 B tile (4 units each).
struct TileRegs { float a0[4], a1[4], b0[4], b1[4]; };

// Global -> registers. tid must have known range [0,255] so the unrolled
// guards fold away.
static __device__ __forceinline__ void load_tile(
    const float* __restrict__ X, const float* __restrict__ W,
    int N, int K, int blockM, int blockN, int k0, int tid, TileRegs& t)
{
    const bool fullK = (k0 + 32) <= K;
    const bool fullN = (blockN + 64) <= N;
    if (fullK && fullN) {
        #pragma unroll
        for (int j = 0; j < 4; ++j) {
            int u = tid + j * 256;
            const float* srcA = X + (size_t)(blockM + (u >> 4)) * K + k0 + (u & 15) * 2;
            t.a0[j] = srcA[0];
            t.a1[j] = srcA[1];
            const float* srcB = W + (size_t)(k0 + (u >> 6) * 2) * N + blockN + (u & 63);
            t.b0[j] = srcB[0];
            t.b1[j] = srcB[N];
        }
    } else {
        #pragma unroll
        for (int j = 0; j < 4; ++j) {
            int u = tid + j * 256;
            int gcA = k0 + (u & 15) * 2;
            const float* rowA = X + (size_t)(blockM + (u >> 4)) * K;
            t.a0[j] = (gcA     < K) ? rowA[gcA]     : 0.0f;
            t.a1[j] = (gcA + 1 < K) ? rowA[gcA + 1] : 0.0f;
            int grB = k0 + (u >> 6) * 2, gcB = blockN + (u & 63);
            t.b0[j] = (grB     < K && gcB < N) ? W[(size_t)grB * N + gcB]       : 0.0f;
            t.b1[j] = (grB + 1 < K && gcB < N) ? W[(size_t)(grB + 1) * N + gcB] : 0.0f;
        }
    }
}

// Registers -> LDS in WMMA fragment layout (ISA 7.12.2):
//   A 16x32 f16: lane(l<16) row=l holds K {0..7,16..23}; lane(l+16) holds
//                K {8..15,24..31}  -> sAf[mtile][lane][8 x uint(2 halves)]
//   B 32x16 f16: lane(l<16) col=l holds K {0..7,16..23}; lane(l+16) K {8..15,24..31}
//                -> sBf[ntile][lane][8 x uint]
static __device__ __forceinline__ void store_tile(
    unsigned (*sAf)[32][8], unsigned (*sBf)[32][8], int tid, const TileRegs& t)
{
    #pragma unroll
    for (int j = 0; j < 4; ++j) {
        int u  = tid + j * 256;
        int r  = u >> 4;                 // A block row 0..63
        int c0 = (u & 15) * 2;           // A even K offset 0..30
        int hf = (c0 >> 3) & 1;
        int e  = (c0 & 7) + ((c0 >> 4) << 3);
        sAf[r >> 4][hf * 16 + (r & 15)][e >> 1] = pack2h(t.a0[j], t.a1[j]);
        int c  = u & 63;                 // B N offset 0..63
        int q  = u >> 6;                 // B K pair 0..15
        sBf[c >> 4][((q & 4) << 2) + (c & 15)][(q & 3) + ((q >> 3) << 2)] = pack2h(t.b0[j], t.b1[j]);
    }
}

// ---------------------------------------------------------------------------
// GEMM: Y[M,N] = X[M,K] * W[K,N] + bias[N]
// Block tile 64(M) x 64(N), 256 threads = 8 waves in a 4(M) x 2(N) grid.
// Each wave owns two 16x16 output tiles (one A fragment, two B fragments,
// two v_wmma per K-step of 32). M is a multiple of 64 for every layer in
// this network, so only K/N edges are guarded (uniformly per block).
// ---------------------------------------------------------------------------
__global__ __launch_bounds__(256)
void gemm_bias_wmma(const float* __restrict__ X, const float* __restrict__ W,
                    const float* __restrict__ bias, float* __restrict__ Y,
                    int M, int N, int K)
{
    __shared__ alignas(32) unsigned sAf[4][32][8];
    __shared__ alignas(32) unsigned sBf[4][32][8];

    const int tid  = threadIdx.x & 255;  // bounded range -> guard-free unrolls
    const int wave = tid >> 5;
    const int lane = tid & 31;
    const int wm   = wave & 3;           // wave's M tile (0..3)
    const int wng  = wave >> 2;          // wave's N tile pair (0..1)
    const int blockM = blockIdx.x * 64;
    const int blockN = blockIdx.y * 64;

    v8f acc0 = {};
    v8f acc1 = {};

    TileRegs t;
    load_tile(X, W, N, K, blockM, blockN, 0, tid, t);

    for (int k0 = 0; k0 < K; k0 += 32) {
        store_tile(sAf, sBf, tid, t);
        __syncthreads();

        // Pipelined fetch of tile k0+32 while tile k0 is consumed from LDS.
        if (k0 + 32 < K)
            load_tile(X, W, N, K, blockM, blockN, k0 + 32, tid, t);
        if (k0 + 64 < K) {   // and nudge tile k0+64 toward the caches
            __builtin_prefetch(X + (size_t)(blockM + (tid >> 2)) * K + k0 + 64, 0, 1);
            __builtin_prefetch(W + (size_t)(k0 + 64 + (tid >> 3)) * N + blockN, 0, 1);
        }

        // contiguous 32-byte fragment loads -> 2x ds_load_b128 each
        v16h a  = *reinterpret_cast<const v16h*>(&sAf[wm][lane][0]);
        v16h b0 = *reinterpret_cast<const v16h*>(&sBf[wng * 2 + 0][lane][0]);
        v16h b1 = *reinterpret_cast<const v16h*>(&sBf[wng * 2 + 1][lane][0]);
        acc0 = __builtin_amdgcn_wmma_f32_16x16x32_f16(
                   false, a, false, b0, (short)0, acc0, false, false);
        acc1 = __builtin_amdgcn_wmma_f32_16x16x32_f16(
                   false, a, false, b1, (short)0, acc1, false, false);
        __syncthreads();
    }

    // C/D layout: acc[r] -> row (half*8 + r), col = lane&15
    const int half = lane >> 4;
    const int l    = lane & 15;
    const int rowBase = blockM + wm * 16 + half * 8;   // M % 64 == 0, no row guard
    #pragma unroll
    for (int tsel = 0; tsel < 2; ++tsel) {
        int col = blockN + (wng * 2 + tsel) * 16 + l;
        if (col < N) {
            float bv = bias ? bias[col] : 0.0f;
            v8f acc = tsel ? acc1 : acc0;
            #pragma unroll
            for (int r = 0; r < 8; ++r)
                Y[(size_t)(rowBase + r) * N + col] = acc[r] + bv;
        }
    }
}

// ---------------------------------------------------------------------------
// BatchNorm statistics: per-channel mean/var over M rows -> fused scale/shift
// ---------------------------------------------------------------------------
__global__ __launch_bounds__(256)
void bn_stats(const float* __restrict__ Y, int M, int N,
              const float* __restrict__ g, const float* __restrict__ be,
              float* __restrict__ scale, float* __restrict__ shift)
{
    const int c = blockIdx.x;
    __shared__ float s_sum[256], s_sq[256];
    float sum = 0.0f, sq = 0.0f;
    for (int m = threadIdx.x; m < M; m += 256) {
        float v = Y[(size_t)m * N + c];
        sum += v; sq += v * v;
    }
    s_sum[threadIdx.x] = sum; s_sq[threadIdx.x] = sq;
    __syncthreads();
    for (int s = 128; s > 0; s >>= 1) {
        if (threadIdx.x < s) {
            s_sum[threadIdx.x] += s_sum[threadIdx.x + s];
            s_sq[threadIdx.x]  += s_sq[threadIdx.x + s];
        }
        __syncthreads();
    }
    if (threadIdx.x == 0) {
        float mean = s_sum[0] / (float)M;
        float var  = s_sq[0] / (float)M - mean * mean;
        float sc   = g[c] * rsqrtf(var + 1e-5f);
        scale[c] = sc;
        shift[c] = be[c] - mean * sc;
    }
}

__global__ __launch_bounds__(256)
void bn_relu(float* __restrict__ Y, int M, int N,
             const float* __restrict__ scale, const float* __restrict__ shift)
{
    size_t i = (size_t)blockIdx.x * 256 + threadIdx.x;
    if (i < (size_t)M * N) {
        int c = (int)(i % (size_t)N);
        float v = Y[i] * scale[c] + shift[c];
        Y[i] = v > 0.0f ? v : 0.0f;
    }
}

// ---------------------------------------------------------------------------
// Farthest point sampling. One block per batch; dist[] lives in LDS.
// ---------------------------------------------------------------------------
__global__ __launch_bounds__(256)
void fps_kernel(const float* __restrict__ xyz, int N, int npoint,
                int* __restrict__ idx_out, float* __restrict__ newxyz)
{
    const int b = blockIdx.x;
    const float* p = xyz + (size_t)b * N * 3;
    extern __shared__ char fsm[];
    float* dist = (float*)fsm;                 // N floats
    float* rmax = dist + N;                    // 256
    int*   rarg = (int*)(rmax + 256);          // 256
    for (int i = threadIdx.x; i < N; i += 256) dist[i] = 1e10f;
    __syncthreads();

    int far = 0;
    for (int it = 0; it < npoint; ++it) {
        if (threadIdx.x == 0) {
            idx_out[b * npoint + it] = far;
            newxyz[((size_t)b * npoint + it) * 3 + 0] = p[far * 3 + 0];
            newxyz[((size_t)b * npoint + it) * 3 + 1] = p[far * 3 + 1];
            newxyz[((size_t)b * npoint + it) * 3 + 2] = p[far * 3 + 2];
        }
        float cx = p[far * 3 + 0], cy = p[far * 3 + 1], cz = p[far * 3 + 2];
        float best = -1.0f; int bestI = 0;
        for (int i = threadIdx.x; i < N; i += 256) {
            float dx = p[i * 3 + 0] - cx;
            float dy = p[i * 3 + 1] - cy;
            float dz = p[i * 3 + 2] - cz;
            float d  = dx * dx + dy * dy + dz * dz;
            float nd = fminf(dist[i], d);
            dist[i] = nd;
            if (nd > best) { best = nd; bestI = i; }
        }
        rmax[threadIdx.x] = best; rarg[threadIdx.x] = bestI;
        __syncthreads();
        for (int s = 128; s > 0; s >>= 1) {
            if (threadIdx.x < s && rmax[threadIdx.x + s] > rmax[threadIdx.x]) {
                rmax[threadIdx.x] = rmax[threadIdx.x + s];
                rarg[threadIdx.x] = rarg[threadIdx.x + s];
            }
            __syncthreads();
        }
        far = rarg[0];
        __syncthreads();
    }
}

// ---------------------------------------------------------------------------
// Ball query: one wave32 per query; __ballot + popc-prefix appends indices in
// ascending order (== reference sort-then-truncate semantics).
// ---------------------------------------------------------------------------
__global__ __launch_bounds__(256)
void ball_query(const float* __restrict__ xyz, const float* __restrict__ new_xyz,
                int N, int S, int nsample, float r2, int BS, int* __restrict__ out)
{
    int gw   = blockIdx.x * 8 + (threadIdx.x >> 5);
    int lane = threadIdx.x & 31;
    if (gw >= BS) return;
    int b = gw / S;
    const float qx = new_xyz[(size_t)gw * 3 + 0];
    const float qy = new_xyz[(size_t)gw * 3 + 1];
    const float qz = new_xyz[(size_t)gw * 3 + 2];
    const float* px = xyz + (size_t)b * N * 3;
    int* dst = out + (size_t)gw * nsample;

    int cnt = 0, first = 0; bool haveFirst = false;
    for (int base = 0; base < N && cnt < nsample; base += 32) {
        int i = base + lane;
        bool in = false;
        if (i < N) {
            float dx = qx - px[i * 3 + 0];
            float dy = qy - px[i * 3 + 1];
            float dz = qz - px[i * 3 + 2];
            in = (dx * dx + dy * dy + dz * dz) <= r2;
        }
        unsigned mm = (unsigned)__ballot(in);
        if (!haveFirst && mm) { first = base + (__ffs((int)mm) - 1); haveFirst = true; }
        int pre = __popc(mm & ((1u << lane) - 1u));
        if (in && cnt + pre < nsample) dst[cnt + pre] = i;
        cnt += __popc(mm);
    }
    if (cnt > nsample) cnt = nsample;
    for (int j = cnt + lane; j < nsample; j += 32) dst[j] = first;
}

// ---------------------------------------------------------------------------
// Grouping: gfeat[q,k,:] = concat(feats[b, idx], xyz[b, idx] - new_xyz[q])
// ---------------------------------------------------------------------------
__global__ __launch_bounds__(256)
void group_kernel(const float* __restrict__ xyz, const float* __restrict__ feats, int C,
                  const float* __restrict__ new_xyz, const int* __restrict__ idx,
                  int BS, int S, int N, int K, float* __restrict__ out)
{
    int t = blockIdx.x * 256 + threadIdx.x;
    if (t >= BS * K) return;
    int q = t / K;
    int b = q / S;
    int gi = idx[t];
    const float* src = feats + ((size_t)b * N + gi) * C;
    float* dst = out + (size_t)t * (C + 3);
    for (int c = 0; c < C; ++c) dst[c] = src[c];
    const float* pp = xyz + ((size_t)b * N + gi) * 3;
    const float* qq = new_xyz + (size_t)q * 3;
    dst[C + 0] = pp[0] - qq[0];
    dst[C + 1] = pp[1] - qq[1];
    dst[C + 2] = pp[2] - qq[2];
}

// Max over group dim K: in [Q,K,C] -> out[q, outOff + c] with row stride outStride
__global__ __launch_bounds__(256)
void maxpool_k(const float* __restrict__ in, int Q, int K, int C,
               float* __restrict__ out, int outStride, int outOff)
{
    int t = blockIdx.x * 256 + threadIdx.x;
    if (t >= Q * C) return;
    int q = t / C, c = t % C;
    float m = -3.4e38f;
    for (int k = 0; k < K; ++k)
        m = fmaxf(m, in[((size_t)q * K + k) * C + c]);
    out[(size_t)q * outStride + outOff + c] = m;
}

// out[m,:] = concat(A[m,0:Ca], B[m,0:Cb])
__global__ __launch_bounds__(256)
void concat2(const float* __restrict__ A, int Ca, const float* __restrict__ Bv, int Cb,
             int Mrows, float* __restrict__ out)
{
    size_t t = (size_t)blockIdx.x * 256 + threadIdx.x;
    int Ct = Ca + Cb;
    if (t >= (size_t)Mrows * Ct) return;
    int m = (int)(t / Ct), c = (int)(t % Ct);
    out[t] = (c < Ca) ? A[(size_t)m * Ca + c] : Bv[(size_t)m * Cb + (c - Ca)];
}

// out[b*S+s,:] = concat(A[b*S+s,0:Ca], B[b,0:Cb])  (S==1 broadcast interp path)
__global__ __launch_bounds__(256)
void concat_broadcast(const float* __restrict__ A, int Ca, const float* __restrict__ Bv, int Cb,
                      int Bn, int S, float* __restrict__ out)
{
    size_t t = (size_t)blockIdx.x * 256 + threadIdx.x;
    int Ct = Ca + Cb;
    if (t >= (size_t)Bn * S * Ct) return;
    int m = (int)(t / Ct), c = (int)(t % Ct);
    int b = m / S;
    out[t] = (c < Ca) ? A[(size_t)m * Ca + c] : Bv[(size_t)b * Cb + (c - Ca)];
}

// 3-NN inverse-distance interpolation fused with concat:
// out[t,:] = concat(p1[t,0:C1], sum_j w_j * p2[b, nn_j, 0:C2])
__global__ __launch_bounds__(256)
void three_interp_concat(const float* __restrict__ xyz1, int N1,
                         const float* __restrict__ xyz2, int N2,
                         const float* __restrict__ p1, int C1,
                         const float* __restrict__ p2, int C2,
                         int Bn, float* __restrict__ out)
{
    int t = blockIdx.x * 256 + threadIdx.x;
    if (t >= Bn * N1) return;
    int b = t / N1;
    const float qx = xyz1[(size_t)t * 3 + 0];
    const float qy = xyz1[(size_t)t * 3 + 1];
    const float qz = xyz1[(size_t)t * 3 + 2];
    const float* b2 = xyz2 + (size_t)b * N2 * 3;
    float d0 = 1e30f, d1 = 1e30f, d2 = 1e30f;
    int   i0 = 0, i1 = 0, i2 = 0;
    for (int j = 0; j < N2; ++j) {
        float dx = qx - b2[j * 3 + 0];
        float dy = qy - b2[j * 3 + 1];
        float dz = qz - b2[j * 3 + 2];
        float d  = dx * dx + dy * dy + dz * dz;
        if (d < d0)      { d2 = d1; i2 = i1; d1 = d0; i1 = i0; d0 = d; i0 = j; }
        else if (d < d1) { d2 = d1; i2 = i1; d1 = d; i1 = j; }
        else if (d < d2) { d2 = d; i2 = j; }
    }
    float w0 = 1.0f / (d0 + 1e-8f), w1 = 1.0f / (d1 + 1e-8f), w2 = 1.0f / (d2 + 1e-8f);
    float ws = w0 + w1 + w2; w0 /= ws; w1 /= ws; w2 /= ws;
    float* dst = out + (size_t)t * (C1 + C2);
    const float* f1 = p1 + (size_t)t * C1;
    for (int c = 0; c < C1; ++c) dst[c] = f1[c];
    const float* fb = p2 + (size_t)b * N2 * C2;
    for (int c = 0; c < C2; ++c)
        dst[C1 + c] = w0 * fb[(size_t)i0 * C2 + c]
                    + w1 * fb[(size_t)i1 * C2 + c]
                    + w2 * fb[(size_t)i2 * C2 + c];
}

__global__ __launch_bounds__(256)
void sigmoid_k(const float* __restrict__ in, float* __restrict__ out, size_t n)
{
    size_t i = (size_t)blockIdx.x * 256 + threadIdx.x;
    if (i < n) out[i] = 1.0f / (1.0f + expf(-in[i]));
}

// ---------------------------------------------------------------------------
// Host orchestration
// ---------------------------------------------------------------------------
struct Layer { const float *W, *b, *g, *be; int din, dout; };

extern "C" void kernel_launch(void* const* d_in, const int* in_sizes, int n_in,
                              void* d_out, int out_size, void* d_ws, size_t ws_size,
                              hipStream_t stream)
{
    (void)in_sizes; (void)n_in; (void)out_size; (void)ws_size;
    const int B = 16, N0 = 4096;

    int cur = 0;
    const float* pc = (const float*)d_in[cur++];   // [B, 4096, 3]

    auto L = [&](int din, int dout) -> Layer {
        Layer x;
        x.W  = (const float*)d_in[cur++];
        x.b  = (const float*)d_in[cur++];
        x.g  = (const float*)d_in[cur++];
        x.be = (const float*)d_in[cur++];
        x.din = din; x.dout = dout; return x;
    };
    // Insertion order of setup_inputs(): sa1, sa2, sa3, fp3, fp2, fp1, head, out.
    Layer sa1b0[3] = { L(6, 32),    L(32, 32),   L(32, 64)   };
    Layer sa1b1[3] = { L(6, 64),    L(64, 64),   L(64, 128)  };
    Layer sa1b2[3] = { L(6, 64),    L(64, 96),   L(96, 128)  };
    Layer sa2b0[3] = { L(323, 128), L(128, 128), L(128, 256) };
    Layer sa2b1[3] = { L(323, 128), L(128, 196), L(196, 256) };
    Layer sa3l[3]  = { L(515, 256), L(256, 512), L(512, 1024)};
    Layer fp3l[2]  = { L(1536, 256), L(256, 256) };
    Layer fp2l[2]  = { L(576, 256),  L(256, 128) };
    Layer fp1l[2]  = { L(131, 128),  L(128, 128) };
    Layer headl[1] = { L(128, 128) };
    const float* outW = (const float*)d_in[cur++];  // [128, 6]
    const float* outB = (const float*)d_in[cur++];  // [6]

    // Workspace bump allocator
    size_t off = 0;
    auto alloc = [&](size_t bytes) -> void* {
        void* p = (char*)d_ws + off;
        off += (bytes + 255) & ~(size_t)255;
        return p;
    };
    const size_t M_MAX = (size_t)B * 512 * 128;                 // 1,048,576 rows
    float* ping    = (float*)alloc(M_MAX * 128 * sizeof(float));
    float* pong    = (float*)alloc(M_MAX * 128 * sizeof(float));
    float* gbuf    = (float*)alloc((size_t)B * 128 * 128 * 323 * sizeof(float));
    int*   l1_idx  = (int*)  alloc((size_t)B * 512 * sizeof(int));
    float* l1_xyz  = (float*)alloc((size_t)B * 512 * 3 * sizeof(float));
    float* l1_p    = (float*)alloc((size_t)B * 512 * 320 * sizeof(float));
    int*   bidx    = (int*)  alloc((size_t)B * 512 * 128 * sizeof(int));
    int*   l2_idx  = (int*)  alloc((size_t)B * 128 * sizeof(int));
    float* l2_xyz  = (float*)alloc((size_t)B * 128 * 3 * sizeof(float));
    float* l2_p    = (float*)alloc((size_t)B * 128 * 512 * sizeof(float));
    float* scale   = (float*)alloc(1024 * sizeof(float));
    float* shift   = (float*)alloc(1024 * sizeof(float));
    float* l2p_new = (float*)alloc((size_t)B * 128 * 256 * sizeof(float));
    float* l1p_new = (float*)alloc((size_t)B * 512 * 128 * sizeof(float));
    float* l0_p    = (float*)alloc((size_t)B * 4096 * 128 * sizeof(float));
    float* outtmp  = (float*)alloc((size_t)B * 4096 * 6 * sizeof(float));

    auto run_gemm = [&](const float* X, const Layer& Ly, float* Y, int M) {
        dim3 grid((unsigned)((M + 63) / 64), (unsigned)((Ly.dout + 63) / 64));
        gemm_bias_wmma<<<grid, 256, 0, stream>>>(X, Ly.W, Ly.b, Y, M, Ly.dout, Ly.din);
    };
    auto run_bnrelu = [&](float* Y, const Layer& Ly, int M) {
        bn_stats<<<Ly.dout, 256, 0, stream>>>(Y, M, Ly.dout, Ly.g, Ly.be, scale, shift);
        size_t tot = (size_t)M * Ly.dout;
        bn_relu<<<(unsigned)((tot + 255) / 256), 256, 0, stream>>>(Y, M, Ly.dout, scale, shift);
    };
    auto run_chain = [&](const Layer* Ls, int n, const float* X0, int M) -> float* {
        const float* X = X0;
        float* Y = ping;
        for (int i = 0; i < n; ++i) {
            run_gemm(X, Ls[i], Y, M);
            run_bnrelu(Y, Ls[i], M);
            X = Y;
            Y = (Y == ping) ? pong : ping;
        }
        return (float*)X;
    };

    const size_t fps_shmem = (size_t)N0 * sizeof(float) + 256 * sizeof(float) + 256 * sizeof(int);

    // ---------------- SA1 (MSG, npoint=512) ----------------
    fps_kernel<<<B, 256, fps_shmem, stream>>>(pc, N0, 512, l1_idx, l1_xyz);
    {
        const Layer* brs[3] = { sa1b0, sa1b1, sa1b2 };
        const int   Ks[3]   = { 32, 64, 128 };
        const float rs[3]   = { 0.1f, 0.2f, 0.4f };
        const int   coff[3] = { 0, 64, 192 };
        const int S = 512, BS = B * S;
        for (int br = 0; br < 3; ++br) {
            int K = Ks[br];
            ball_query<<<(unsigned)((BS + 7) / 8), 256, 0, stream>>>(
                pc, l1_xyz, N0, S, K, rs[br] * rs[br], BS, bidx);
            int QK = BS * K;
            group_kernel<<<(unsigned)((QK + 255) / 256), 256, 0, stream>>>(
                pc, pc, 3, l1_xyz, bidx, BS, S, N0, K, gbuf);
            float* h = run_chain(brs[br], 3, gbuf, QK);
            int C = brs[br][2].dout;
            maxpool_k<<<(unsigned)((BS * C + 255) / 256), 256, 0, stream>>>(
                h, BS, K, C, l1_p, 320, coff[br]);
        }
    }

    // ---------------- SA2 (MSG, npoint=128) ----------------
    fps_kernel<<<B, 256, fps_shmem, stream>>>(l1_xyz, 512, 128, l2_idx, l2_xyz);
    {
        const Layer* brs[2] = { sa2b0, sa2b1 };
        const int   Ks[2]   = { 64, 128 };
        const float rs[2]   = { 0.4f, 0.8f };
        const int   coff[2] = { 0, 256 };
        const int S = 128, BS = B * S, N = 512;
        for (int br = 0; br < 2; ++br) {
            int K = Ks[br];
            ball_query<<<(unsigned)((BS + 7) / 8), 256, 0, stream>>>(
                l1_xyz, l2_xyz, N, S, K, rs[br] * rs[br], BS, bidx);
            int QK = BS * K;
            group_kernel<<<(unsigned)((QK + 255) / 256), 256, 0, stream>>>(
                l1_xyz, l1_p, 320, l2_xyz, bidx, BS, S, N, K, gbuf);
            float* h = run_chain(brs[br], 3, gbuf, QK);
            int C = brs[br][2].dout;
            maxpool_k<<<(unsigned)((BS * C + 255) / 256), 256, 0, stream>>>(
                h, BS, K, C, l2_p, 512, coff[br]);
        }
    }

    // ---------------- SA3 (group all) ----------------
    float* l3_p = (float*)d_out + (size_t)B * N0 * 6;   // second return value slot
    {
        int M3 = B * 128;
        size_t tot = (size_t)M3 * 515;
        concat2<<<(unsigned)((tot + 255) / 256), 256, 0, stream>>>(l2_xyz, 3, l2_p, 512, M3, gbuf);
        float* h = run_chain(sa3l, 3, gbuf, M3);
        maxpool_k<<<(unsigned)((B * 1024 + 255) / 256), 256, 0, stream>>>(
            h, B, 128, 1024, l3_p, 1024, 0);
    }

    // ---------------- FP3 (S==1 broadcast) ----------------
    {
        int M3 = B * 128;
        size_t tot = (size_t)M3 * 1536;
        concat_broadcast<<<(unsigned)((tot + 255) / 256), 256, 0, stream>>>(
            l2_p, 512, l3_p, 1024, B, 128, gbuf);
        float* h = run_chain(fp3l, 2, gbuf, M3);
        hipMemcpyAsync(l2p_new, h, (size_t)M3 * 256 * sizeof(float),
                       hipMemcpyDeviceToDevice, stream);
    }

    // ---------------- FP2 ----------------
    {
        int M2 = B * 512;
        three_interp_concat<<<(unsigned)((M2 + 255) / 256), 256, 0, stream>>>(
            l1_xyz, 512, l2_xyz, 128, l1_p, 320, l2p_new, 256, B, gbuf);
        float* h = run_chain(fp2l, 2, gbuf, M2);
        hipMemcpyAsync(l1p_new, h, (size_t)M2 * 128 * sizeof(float),
                       hipMemcpyDeviceToDevice, stream);
    }

    // ---------------- FP1 ----------------
    {
        int M1 = B * N0;
        three_interp_concat<<<(unsigned)((M1 + 255) / 256), 256, 0, stream>>>(
            pc, N0, l1_xyz, 512, pc, 3, l1p_new, 128, B, gbuf);
        float* h = run_chain(fp1l, 2, gbuf, M1);
        hipMemcpyAsync(l0_p, h, (size_t)M1 * 128 * sizeof(float),
                       hipMemcpyDeviceToDevice, stream);
    }

    // ---------------- Head + sigmoid output ----------------
    {
        int M1 = B * N0;
        float* hh = run_chain(headl, 1, l0_p, M1);
        dim3 grid((unsigned)((M1 + 63) / 64), 1);
        gemm_bias_wmma<<<grid, 256, 0, stream>>>(hh, outW, outB, outtmp, M1, 6, 128);
        size_t tot = (size_t)M1 * 6;
        sigmoid_k<<<(unsigned)((tot + 255) / 256), 256, 0, stream>>>(outtmp, (float*)d_out, tot);
    }
}
